// MutiLlamaAttention_33775622816048
// MI455X (gfx1250) — matrix operand order repaired and checked by reading, not compiled, not run on hardware
//
#include <hip/hip_runtime.h>
#include <hip/hip_bf16.h>
#include <math.h>

#define S_LEN 2048
#define HID_D 2048
#define NH    32
#define NKV   8
#define HD    64
#define CTX   (S_LEN - 10)
#define NEGBIG -1000000000.0f

typedef __attribute__((ext_vector_type(16))) __bf16 v16bf;
typedef __attribute__((ext_vector_type(8)))  __bf16 v8bf;
typedef __attribute__((ext_vector_type(8)))  float  v8f;
typedef __attribute__((ext_vector_type(4)))  unsigned int v4u;
typedef __attribute__((ext_vector_type(8)))  int v8i;
typedef __attribute__((ext_vector_type(4)))  int v4i;

struct bf2 { __bf16 h, l; };
__device__ __forceinline__ bf2 bsplit(float x) {
    bf2 r;
    r.h = (__bf16)x;
    r.l = (__bf16)(x - (float)r.h);
    return r;
}

__device__ __forceinline__ v8bf ld8(const __bf16* p) { return *(const v8bf*)p; }
__device__ __forceinline__ v16bf cat8(v8bf a, v8bf b) {
    return __builtin_shufflevector(a, b, 0,1,2,3,4,5,6,7,8,9,10,11,12,13,14,15);
}

// 3-term bf16 emulated-f32 MAC:  C += Ah*Bh + Ah*Bl + Al*Bh
__device__ __forceinline__ v8f wmma3(v16bf ah, v16bf al, v16bf bh, v16bf bl, v8f c) {
    c = __builtin_amdgcn_wmma_f32_16x16x32_bf16(false, ah, false, bh, (short)0, c, false, false);
    c = __builtin_amdgcn_wmma_f32_16x16x32_bf16(false, ah, false, bl, (short)0, c, false, false);
    c = __builtin_amdgcn_wmma_f32_16x16x32_bf16(false, al, false, bh, (short)0, c, false, false);
    return c;
}

// A-operand (16x32 bf16) K-offset for half h (0..15), lane-half group hi (0/1)
__device__ __forceinline__ int amap(int h, int hi) {
    return h + hi * 16;
}

__device__ __forceinline__ unsigned lds_off_of(const void* p) {
    // flat LDS address: low 32 bits are the LDS byte offset (ISA 10.2 aperture rules)
    return (unsigned)(unsigned long long)p;
}

// ---------------------------------------------------------------------------
// TDM: stage a 64-row x 32-col bf16 tile (row stride K elements) into LDS,
// padding 4 DWORDs after every 16 DWORDs (64B row -> 80B LDS row stride).
// D# per CDNA5 ISA sec 8.3/8.4. Issued per-wave (EXEC ignored by TDM).
// ---------------------------------------------------------------------------
__device__ __forceinline__ void tdm_load_panel(const __bf16* gsrc, unsigned lds_byte_off, int K) {
    unsigned long long ga = (unsigned long long)(const void*)gsrc;
    v4u g0;
    g0[0] = 1u;                                            // count=1, user D#
    g0[1] = lds_byte_off;                                  // lds_addr
    g0[2] = (unsigned)(ga & 0xFFFFFFFFu);                  // global_addr lo
    g0[3] = (unsigned)((ga >> 32) & 0x01FFFFFFu)           // global_addr hi (57b)
          | 0x80000000u;                                   // type=2 ("image")
    v8i g1;
    g1[0] = (int)(0x00010000u                              // data_size=1 -> 2 bytes
                | (1u << 20)                               // pad_enable
                | (3u << 22)                               // pad_interval: 16 DWORDs
                | (3u << 25));                             // pad_amount: 4 DWORDs
    g1[1] = (int)(((unsigned)K & 0xFFFFu) << 16);          // tensor_dim0[15:0]
    g1[2] = (int)(((unsigned)K >> 16) | (64u << 16));      // tensor_dim0[31:16] | tensor_dim1[15:0]=64
    g1[3] = (int)(32u << 16);                              // tensor_dim1 hi=0 | tile_dim0=32
    g1[4] = 64;                                            // tile_dim1=64, tile_dim2=0
    g1[5] = K;                                             // tensor_dim0_stride lo32 (elements)
    g1[6] = 0;                                             // stride hi | dim1_stride lo
    g1[7] = 0;
    v4i g2; g2[0] = 0; g2[1] = 0; g2[2] = 0; g2[3] = 0;    // unused (2D)
    v4i g3; g3[0] = 0; g3[1] = 0; g3[2] = 0; g3[3] = 0;
    v8i g4; g4[0] = 0; g4[1] = 0; g4[2] = 0; g4[3] = 0;    // unused trailing group
    g4[4] = 0; g4[5] = 0; g4[6] = 0; g4[7] = 0;
    __builtin_amdgcn_tensor_load_to_lds(g0, g1, g2, g3, g4, 0);
}

// ---------------------------------------------------------------------------
// Elementwise f32 -> bf16 hi/lo split (same layout).
// ---------------------------------------------------------------------------
__global__ void split_kernel(const float* __restrict__ src, __bf16* __restrict__ hi,
                             __bf16* __restrict__ lo, int n) {
    int i = blockIdx.x * blockDim.x + threadIdx.x;
    if (i >= n) return;
    bf2 t = bsplit(src[i]);
    hi[i] = t.h;
    lo[i] = t.l;
}

// ---------------------------------------------------------------------------
// V split + transpose: v[kv][s][64] (f32) -> th/tl[kv][64][S] (bf16)
// ---------------------------------------------------------------------------
__global__ void vsplit_kernel(const float* __restrict__ v,
                              __bf16* __restrict__ th, __bf16* __restrict__ tl) {
    int i = blockIdx.x * blockDim.x + threadIdx.x;   // over NKV*S*64 = 2^20
    if (i >= NKV * S_LEN * HD) return;
    int d = i & 63;
    int s = (i >> 6) & (S_LEN - 1);
    int h = i >> 17;                                  // S*64 = 2^17
    bf2 t = bsplit(v[i]);
    size_t o = ((size_t)h * HD + d) * S_LEN + s;
    th[o] = t.h;
    tl[o] = t.l;
}

// ---------------------------------------------------------------------------
// Dual RoPE with direct bf16 hi/lo split output.
// ---------------------------------------------------------------------------
__global__ void rope_split_kernel(const float* __restrict__ src,
                                  __bf16* __restrict__ obh, __bf16* __restrict__ obl,
                                  __bf16* __restrict__ onh, __bf16* __restrict__ onl,
                                  int nheads) {
    int idx = blockIdx.x * blockDim.x + threadIdx.x;
    if (idx >= nheads * S_LEN * 32) return;
    int d = idx & 31;
    int s = (idx >> 5) & (S_LEN - 1);
    int h = idx >> 16;                     // 32*2048 = 2^16
    size_t base = (size_t)h * S_LEN * HD + (size_t)s * HD;
    float x1 = src[base + d];
    float x2 = src[base + d + 32];
    float invf = __expf(-(float)d * (9.210340371976184f / 32.0f)); // 10000^(-d/32)
    float ab = (float)s * invf;            // boost: t = pos
    float an = ((float)s * 0.25f) * invf;  // narrow: t = pos / 4
    float sb, cb, sn, cn;
    __sincosf(ab, &sb, &cb);
    __sincosf(an, &sn, &cn);
    bf2 b1 = bsplit(x1 * cb - x2 * sb);
    bf2 b2 = bsplit(x2 * cb + x1 * sb);
    bf2 n1 = bsplit(x1 * cn - x2 * sn);
    bf2 n2 = bsplit(x2 * cn + x1 * sn);
    obh[base + d] = b1.h;  obl[base + d] = b1.l;
    obh[base + d + 32] = b2.h;  obl[base + d + 32] = b2.l;
    onh[base + d] = n1.h;  onl[base + d] = n1.l;
    onh[base + d + 32] = n2.h;  onl[base + d + 32] = n2.l;
}

// ---------------------------------------------------------------------------
// C[M,N] = A[M,K] @ W[N,K]^T, pre-split bf16 operands, f32 acc.
// 4 waves/block: wave w owns 16 M-rows; the 64xK W-panel is shared through
// LDS, staged by the Tensor Data Mover (double-buffered on TENSORcnt).
// ---------------------------------------------------------------------------
#define WP_STRIDE 40   // 32 elements + 8 pad (80B rows -> conflict-free b128)
__global__ __launch_bounds__(128, 1)
void gemm_bf(const __bf16* __restrict__ Ah, const __bf16* __restrict__ Al,
             const __bf16* __restrict__ Wh, const __bf16* __restrict__ Wl,
             float* __restrict__ C, int M, int N, int K, int head_major) {
    __shared__ __bf16 panel[2][2][64 * WP_STRIDE];   // [buf][hi/lo][64 rows x 40]
    const int wave = threadIdx.x >> 5;
    const int lane = threadIdx.x & 31;
    const int lo = lane & 15, hi = lane >> 4;
    const int i0 = (blockIdx.y * 4 + wave) * 16;
    const int n0 = blockIdx.x * 64;

    v8f acc[4];
    #pragma unroll
    for (int nt = 0; nt < 4; ++nt)
        #pragma unroll
        for (int r = 0; r < 8; ++r) acc[nt][r] = 0.0f;

    const size_t arow = (size_t)(i0 + lo) * K;
    const size_t wbase = (size_t)n0 * K;
    const int nk = K / 32;

    if (wave == 0) {
        tdm_load_panel(Wh + wbase, lds_off_of(&panel[0][0][0]), K);
        tdm_load_panel(Wl + wbase, lds_off_of(&panel[0][1][0]), K);
    }
    for (int t = 0; t < nk; ++t) {
        const int kk = t * 32;
        const int buf = t & 1;
        if (wave == 0) __builtin_amdgcn_s_wait_tensorcnt(0);   // panel[buf] ready
        __syncthreads();
        if (wave == 0 && t + 1 < nk) {                          // overlap next DMA
            tdm_load_panel(Wh + wbase + kk + 32, lds_off_of(&panel[buf ^ 1][0][0]), K);
            tdm_load_panel(Wl + wbase + kk + 32, lds_off_of(&panel[buf ^ 1][1][0]), K);
        }
        const __bf16* pa = Ah + arow + kk + hi * 16;
        const __bf16* pl = Al + arow + kk + hi * 16;
        v16bf ah = cat8(ld8(pa), ld8(pa + 8));
        v16bf al = cat8(ld8(pl), ld8(pl + 8));
        __builtin_prefetch(pa + 32, 0, 3);
        #pragma unroll
        for (int nt = 0; nt < 4; ++nt) {
            const __bf16* ph_ = &panel[buf][0][(nt * 16 + lo) * WP_STRIDE + hi * 16];
            const __bf16* pw_ = &panel[buf][1][(nt * 16 + lo) * WP_STRIDE + hi * 16];
            v16bf bh = cat8(ld8(ph_), ld8(ph_ + 8));
            v16bf bl = cat8(ld8(pw_), ld8(pw_ + 8));
            acc[nt] = wmma3(ah, al, bh, bl, acc[nt]);
        }
        __syncthreads();
    }
    #pragma unroll
    for (int nt = 0; nt < 4; ++nt)
        #pragma unroll
        for (int r = 0; r < 8; ++r) {
            int row = i0 + r + hi * 8;
            int col = n0 + nt * 16 + lo;
            float v = acc[nt][r];
            if (head_major) {
                int hh = col >> 6, d = col & 63;
                C[(size_t)hh * M * 64 + (size_t)row * 64 + d] = v;
            } else {
                C[(size_t)row * N + col] = v;
            }
        }
}

// ---------------------------------------------------------------------------
// Fused flash attention, boost/narrow merge, all operands pre-split bf16.
// Grid: (S/64, NH); 4 waves/block, one 16-row q-tile per wave.
// ---------------------------------------------------------------------------
__global__ __launch_bounds__(128, 1)
void attn_kernel(const __bf16* __restrict__ qbh_, const __bf16* __restrict__ qbl_,
                 const __bf16* __restrict__ qnh_, const __bf16* __restrict__ qnl_,
                 const __bf16* __restrict__ kbh_, const __bf16* __restrict__ kbl_,
                 const __bf16* __restrict__ knh_, const __bf16* __restrict__ knl_,
                 const __bf16* __restrict__ vth_, const __bf16* __restrict__ vtl_,
                 float* __restrict__ out) {
    __shared__ float pbuf[4][16][33];
    const int wave = threadIdx.x >> 5;
    const int lane = threadIdx.x & 31;
    const int lo = lane & 15, hi = lane >> 4;
    const int head = blockIdx.y;
    const int kvh  = head >> 2;               // GROUPS = 4
    const int i0   = (blockIdx.x * 4 + wave) * 16;

    const size_t qoff = (size_t)head * S_LEN * HD;
    const size_t koff = (size_t)kvh  * S_LEN * HD;
    const size_t voff = (size_t)kvh  * HD * S_LEN;

    v16bf qBh[2], qBl[2], qNh[2], qNl[2];
    #pragma unroll
    for (int ks = 0; ks < 2; ++ks) {
        size_t o = qoff + (size_t)(i0 + lo) * HD + ks * 32 + hi * 16;
        qBh[ks] = cat8(ld8(qbh_ + o), ld8(qbh_ + o + 8));
        qBl[ks] = cat8(ld8(qbl_ + o), ld8(qbl_ + o + 8));
        qNh[ks] = cat8(ld8(qnh_ + o), ld8(qnh_ + o + 8));
        qNl[ks] = cat8(ld8(qnl_ + o), ld8(qnl_ + o + 8));
    }

    float m_r[8], l_r[8];
    v8f acc[4];
    #pragma unroll
    for (int r = 0; r < 8; ++r) { m_r[r] = -INFINITY; l_r[r] = 0.0f; }
    #pragma unroll
    for (int dt = 0; dt < 4; ++dt)
        #pragma unroll
        for (int r = 0; r < 8; ++r) acc[dt][r] = 0.0f;

    const int nj = (i0 + 16 + 31) / 32;       // causal j-blocks of 32
    for (int jb = 0; jb < nj; ++jb) {
        const int j0 = jb * 32;
        float sc[2][8];
        #pragma unroll
        for (int jt = 0; jt < 2; ++jt) {
            v8f big, small;
            #pragma unroll
            for (int r = 0; r < 8; ++r) { big[r] = 0.0f; small[r] = 0.0f; }
            #pragma unroll
            for (int ks = 0; ks < 2; ++ks) {
                size_t o = koff + (size_t)(j0 + jt * 16 + lo) * HD + ks * 32 + hi * 16;
                v16bf bh  = cat8(ld8(kbh_ + o), ld8(kbh_ + o + 8));
                v16bf bl  = cat8(ld8(kbl_ + o), ld8(kbl_ + o + 8));
                v16bf nh_ = cat8(ld8(knh_ + o), ld8(knh_ + o + 8));
                v16bf nl_ = cat8(ld8(knl_ + o), ld8(knl_ + o + 8));
                big   = wmma3(qBh[ks], qBl[ks], bh,  bl,  big);
                small = wmma3(qNh[ks], qNl[ks], nh_, nl_, small);
            }
            #pragma unroll
            for (int r = 0; r < 8; ++r) {
                int i = i0 + r + hi * 8;
                int j = j0 + jt * 16 + lo;
                float bg = big[r]   * 0.125f;      // 1/sqrt(64)
                float sm = small[r] * 0.125f;
                bool causal = (j <= i);
                float v = (causal && (i < CTX)) ? bg : sm;
                if (!causal) v = NEGBIG;
                sc[jt][r] = v;
            }
        }
        #pragma unroll
        for (int r = 0; r < 8; ++r) {
            float vmax = fmaxf(sc[0][r], sc[1][r]);
            #pragma unroll
            for (int msk = 1; msk < 16; msk <<= 1)
                vmax = fmaxf(vmax, __shfl_xor(vmax, msk, 32));
            float mnew  = fmaxf(m_r[r], vmax);
            float alpha = __expf(m_r[r] - mnew);
            float p0 = __expf(sc[0][r] - mnew);
            float p1 = __expf(sc[1][r] - mnew);
            sc[0][r] = p0; sc[1][r] = p1;
            float rs = p0 + p1;
            #pragma unroll
            for (int msk = 1; msk < 16; msk <<= 1)
                rs += __shfl_xor(rs, msk, 32);
            l_r[r] = l_r[r] * alpha + rs;
            m_r[r] = mnew;
            #pragma unroll
            for (int dt = 0; dt < 4; ++dt) acc[dt][r] *= alpha;
        }
        #pragma unroll
        for (int jt = 0; jt < 2; ++jt)
            #pragma unroll
            for (int r = 0; r < 8; ++r)
                pbuf[wave][r + hi * 8][jt * 16 + lo] = sc[jt][r];
        asm volatile("s_wait_dscnt 0" ::: "memory");
        v16bf ph, pl;
        #pragma unroll
        for (int h = 0; h < 16; ++h) {
            bf2 t = bsplit(pbuf[wave][lo][amap(h, hi)]);
            ph[h] = t.h; pl[h] = t.l;
        }
        asm volatile("s_wait_dscnt 0" ::: "memory");
        #pragma unroll
        for (int dt = 0; dt < 4; ++dt) {
            size_t o = voff + (size_t)(dt * 16 + lo) * S_LEN + j0 + hi * 16;
            v16bf vhh = cat8(ld8(vth_ + o), ld8(vth_ + o + 8));
            v16bf vll = cat8(ld8(vtl_ + o), ld8(vtl_ + o + 8));
            acc[dt] = wmma3(ph, pl, vhh, vll, acc[dt]);
        }
    }
    #pragma unroll
    for (int dt = 0; dt < 4; ++dt)
        #pragma unroll
        for (int r = 0; r < 8; ++r) {
            int row = i0 + r + hi * 8;
            int col = head * HD + dt * 16 + lo;
            out[(size_t)row * HID_D + col] = acc[dt][r] / l_r[r];
        }
}

// ---------------------------------------------------------------------------
extern "C" void kernel_launch(void* const* d_in, const int* in_sizes, int n_in,
                              void* d_out, int out_size, void* d_ws, size_t ws_size,
                              hipStream_t stream) {
    const float* hs = (const float*)d_in[0];
    const float* Wq = (const float*)d_in[3];
    const float* Wk = (const float*)d_in[4];
    const float* Wv = (const float*)d_in[5];
    const float* Wo = (const float*)d_in[6];
    float* out = (float*)d_out;

    const size_t QSZ = (size_t)NH  * S_LEN * HD;   // 4M elements
    const size_t KSZ = (size_t)NKV * S_LEN * HD;   // 1M elements
    const size_t HSZ = (size_t)S_LEN * HID_D;      // 4M elements

    char* p = (char*)d_ws;
    auto alloc_f = [&](size_t n) { float*  r = (float*)p;  p += n * sizeof(float);  return r; };
    auto alloc_b = [&](size_t n) { __bf16* r = (__bf16*)p; p += n * sizeof(__bf16); return r; };

    float* qF = alloc_f(QSZ);      // Q proj (head-major); later reused as context
    float* kF = alloc_f(KSZ);
    float* vF = alloc_f(KSZ);
    __bf16* hsH = alloc_b(HSZ);    // hs split; later reused for context split
    __bf16* hsL = alloc_b(HSZ);
    __bf16* WqH = alloc_b(HSZ);  __bf16* WqL = alloc_b(HSZ);
    __bf16* WkH = alloc_b(KSZ);  __bf16* WkL = alloc_b(KSZ);
    __bf16* WvH = alloc_b(KSZ);  __bf16* WvL = alloc_b(KSZ);
    __bf16* WoH = alloc_b(HSZ);  __bf16* WoL = alloc_b(HSZ);
    __bf16* qBh = alloc_b(QSZ);  __bf16* qBl = alloc_b(QSZ);
    __bf16* qNh = alloc_b(QSZ);  __bf16* qNl = alloc_b(QSZ);
    __bf16* kBh = alloc_b(KSZ);  __bf16* kBl = alloc_b(KSZ);
    __bf16* kNh = alloc_b(KSZ);  __bf16* kNl = alloc_b(KSZ);
    __bf16* vTh = alloc_b(KSZ);  __bf16* vTl = alloc_b(KSZ);

    split_kernel<<<(int)(HSZ / 256), 256, 0, stream>>>(hs, hsH, hsL, (int)HSZ);
    split_kernel<<<(int)(HSZ / 256), 256, 0, stream>>>(Wq, WqH, WqL, (int)HSZ);
    split_kernel<<<(int)(KSZ / 256), 256, 0, stream>>>(Wk, WkH, WkL, (int)KSZ);
    split_kernel<<<(int)(KSZ / 256), 256, 0, stream>>>(Wv, WvH, WvL, (int)KSZ);
    split_kernel<<<(int)(HSZ / 256), 256, 0, stream>>>(Wo, WoH, WoL, (int)HSZ);

    dim3 blk(128);
    gemm_bf<<<dim3((NH * HD) / 64,  S_LEN / 64), blk, 0, stream>>>(hsH, hsL, WqH, WqL, qF, S_LEN, NH * HD,  HID_D, 1);
    gemm_bf<<<dim3((NKV * HD) / 64, S_LEN / 64), blk, 0, stream>>>(hsH, hsL, WkH, WkL, kF, S_LEN, NKV * HD, HID_D, 1);
    gemm_bf<<<dim3((NKV * HD) / 64, S_LEN / 64), blk, 0, stream>>>(hsH, hsL, WvH, WvL, vF, S_LEN, NKV * HD, HID_D, 1);

    rope_split_kernel<<<(NH  * S_LEN * 32) / 256, 256, 0, stream>>>(qF, qBh, qBl, qNh, qNl, NH);
    rope_split_kernel<<<(NKV * S_LEN * 32) / 256, 256, 0, stream>>>(kF, kBh, kBl, kNh, kNl, NKV);
    vsplit_kernel<<<(int)(KSZ / 256), 256, 0, stream>>>(vF, vTh, vTl);

    attn_kernel<<<dim3(S_LEN / 64, NH), dim3(128), 0, stream>>>(
        qBh, qBl, qNh, qNl, kBh, kBl, kNh, kNl, vTh, vTl, qF);

    split_kernel<<<(int)(HSZ / 256), 256, 0, stream>>>(qF, hsH, hsL, (int)HSZ);
    gemm_bf<<<dim3(HID_D / 64, S_LEN / 64), blk, 0, stream>>>(hsH, hsL, WoH, WoL, out, S_LEN, HID_D, HID_D, 0);
}
